// Mamba2D_69337952026960
// MI455X (gfx1250) — compile-verified, matrix operand order checked by reading, not run on hardware
//
#include <hip/hip_runtime.h>
#include <hip/hip_bf16.h>
#include <cstdint>
#include <cstddef>

// ---------------------------------------------------------------------------
// Mamba2D for MI455X (gfx1250, wave32).
// GEMMs use v_wmma_f32_16x16x32_f16 (f16 operands, f32 accumulate) with
// 2x2 output tiles per wave and b128 fragment loads.
// Scan is a register-resident recurrence (16 states/thread) fused with the
// u*D skip and SiLU(z) gating.
// ---------------------------------------------------------------------------

typedef __attribute__((ext_vector_type(16))) _Float16 v16h;
typedef __attribute__((ext_vector_type(8)))  _Float16 v8h;
typedef __attribute__((ext_vector_type(8)))  float    v8f;

#define NB   4        // batch
#define NC   256      // channels C
#define NL   4096     // L = H*W
#define DI   512      // d_inner
#define DSN  16       // d_state
#define NM   (NB*NL)  // 16384 rows

// ---------------------------------------------------------------------------
// Fragment loader for the CDNA5 wave32 16-bit A/B layout (ISA 05_wmma 7.12.2).
// Per lane (m/n = lane&15, half = lane>>4) the 16 f16 fragment elements are
// two contiguous 16-byte runs of K:
//   elems 0..7  -> K = 8*half + [0,8)
//   elems 8..15 -> K = 16 + 8*half + [0,8)
// => exactly two global_load_b128 per lane, no fix-up VALU.
// ---------------------------------------------------------------------------
__device__ inline v16h load_frag16(const _Float16* __restrict__ base, int ld, int lane)
{
    const _Float16* rowp =
        base + (size_t)(lane & 15) * (size_t)ld + ((lane >> 4) << 3);
    v8h lo = *(const v8h*)(rowp);
    v8h hi = *(const v8h*)(rowp + 16);
    return __builtin_shufflevector(lo, hi,
                                   0, 1, 2, 3, 4, 5, 6, 7,
                                   8, 9, 10, 11, 12, 13, 14, 15);
}

// ---------------------------------------------------------------------------
// GEMM: D(MxN,f32) = A(MxK,f16) * W(NxK,f16)^T
// block = 256 threads = 8 waves arranged 4(M) x 2(N); each wave computes a
// 32x32 output tile as 2x2 WMMA tiles (4 wmma per K-step, fragments reused).
// grid = (N/64, M/128).
// mode 0: row-major store.  mode 1: scatter into (B, C, L) output layout.
// ---------------------------------------------------------------------------
__global__ void __launch_bounds__(256)
wmma_gemm_f16f32_kernel(const _Float16* __restrict__ A,
                        const _Float16* __restrict__ W,
                        float* __restrict__ Dst,
                        int M, int N, int K, int mode)
{
    const int lane = threadIdx.x & 31;
    const int wave = threadIdx.x >> 5;
    const int row0 = blockIdx.y * 128 + (wave >> 1) * 32;
    const int col0 = blockIdx.x * 64 + (wave & 1) * 32;

    const _Float16* A0 = A + (size_t)row0 * (size_t)K;
    const _Float16* A1 = A + (size_t)(row0 + 16) * (size_t)K;
    const _Float16* W0 = W + (size_t)col0 * (size_t)K;
    const _Float16* W1 = W + (size_t)(col0 + 16) * (size_t)K;

    v8f acc00 = {}, acc01 = {}, acc10 = {}, acc11 = {};

    for (int k = 0; k < K; k += 32) {
        // keep the next A K-slabs warm in near cache (global_prefetch_b8)
        __builtin_prefetch(A0 + k + 128, 0, 3);
        __builtin_prefetch(A1 + k + 128, 0, 3);

        v16h a0 = load_frag16(A0 + k, K, lane);
        v16h a1 = load_frag16(A1 + k, K, lane);
        v16h b0 = load_frag16(W0 + k, K, lane);
        v16h b1 = load_frag16(W1 + k, K, lane);

        acc00 = __builtin_amdgcn_wmma_f32_16x16x32_f16(
                    false, a0, false, b0, (short)0, acc00, false, false);
        acc01 = __builtin_amdgcn_wmma_f32_16x16x32_f16(
                    false, a0, false, b1, (short)0, acc01, false, false);
        acc10 = __builtin_amdgcn_wmma_f32_16x16x32_f16(
                    false, a1, false, b0, (short)0, acc10, false, false);
        acc11 = __builtin_amdgcn_wmma_f32_16x16x32_f16(
                    false, a1, false, b1, (short)0, acc11, false, false);
    }

    // D tile store: f32 C/D layout — VGPR v, half = lane>>4: m = v + 8*half,
    // n = lane&15 within each 16x16 tile.
    const int half = lane >> 4;
    const int nloc = lane & 15;
    if (mode == 0) {
#pragma unroll
        for (int v = 0; v < 8; ++v) {
            const int m0 = row0 + v + 8 * half;
            const int m1 = m0 + 16;
            Dst[(size_t)m0 * (size_t)N + (col0 + nloc)]      = acc00[v];
            Dst[(size_t)m0 * (size_t)N + (col0 + 16 + nloc)] = acc01[v];
            Dst[(size_t)m1 * (size_t)N + (col0 + nloc)]      = acc10[v];
            Dst[(size_t)m1 * (size_t)N + (col0 + 16 + nloc)] = acc11[v];
        }
    } else {
        // out[b, c, l] : row m = b*NL + l, column n = channel c
#pragma unroll
        for (int v = 0; v < 8; ++v) {
            const int m0 = row0 + v + 8 * half;
            const int m1 = m0 + 16;
            const int b0i = m0 >> 12, l0 = m0 & (NL - 1);   // NL = 2^12
            const int b1i = m1 >> 12, l1 = m1 & (NL - 1);
            Dst[(((size_t)(b0i * NC + col0 + nloc))      << 12) + l0] = acc00[v];
            Dst[(((size_t)(b0i * NC + col0 + 16 + nloc)) << 12) + l0] = acc01[v];
            Dst[(((size_t)(b1i * NC + col0 + nloc))      << 12) + l1] = acc10[v];
            Dst[(((size_t)(b1i * NC + col0 + 16 + nloc)) << 12) + l1] = acc11[v];
        }
    }
}

// ---------------------------------------------------------------------------
// f32 -> f16 weight conversion with optional row zero-padding.
// dst has n = dstRows*cols elements; rows >= srcRows are zero.
// ---------------------------------------------------------------------------
__global__ void __launch_bounds__(256)
cvt_pad_f16_kernel(const float* __restrict__ src, _Float16* __restrict__ dst,
                   int srcRows, int cols, int n)
{
    const int i = blockIdx.x * 256 + threadIdx.x;
    if (i >= n) return;
    const int row = i / cols;
    dst[i] = (row < srcRows) ? (_Float16)src[i] : (_Float16)0.0f;
}

// ---------------------------------------------------------------------------
// LayerNorm over C=256 for one (b,l) row per block; writes f16 activations.
// x layout: (B, C, L); xs[b,l,c] = x[b,c,l].
// ---------------------------------------------------------------------------
__global__ void __launch_bounds__(256)
layernorm_f16_kernel(const float* __restrict__ x,
                     const float* __restrict__ gw,
                     const float* __restrict__ gb,
                     _Float16* __restrict__ xn)
{
    __shared__ float s1[256];
    __shared__ float s2[256];
    const int row = blockIdx.x;            // b*NL + l
    const int b   = row >> 12;
    const int l   = row & (NL - 1);
    const int c   = threadIdx.x;

    const float v = x[(((size_t)(b * NC + c)) << 12) + l];
    s1[c] = v;
    s2[c] = v * v;
    __syncthreads();
#pragma unroll
    for (int off = 128; off > 0; off >>= 1) {
        if (c < off) { s1[c] += s1[c + off]; s2[c] += s2[c + off]; }
        __syncthreads();
    }
    const float mu  = s1[0] * (1.0f / NC);
    const float var = s2[0] * (1.0f / NC) - mu * mu;
    const float rs  = rsqrtf(var + 1e-5f);
    xn[(size_t)row * NC + c] = (_Float16)((v - mu) * rs * gw[c] + gb[c]);
}

// ---------------------------------------------------------------------------
// Depthwise causal conv1d (k=4) along L + SiLU. u lives in xz cols [0,512).
// Writes f32 (for the scan) and f16 (for the x_proj GEMM).
// ---------------------------------------------------------------------------
__global__ void __launch_bounds__(256)
conv_silu_kernel(const float* __restrict__ xz,
                 const float* __restrict__ cw,
                 const float* __restrict__ cb,
                 float* __restrict__ u32,
                 _Float16* __restrict__ u16)
{
    const int idx = blockIdx.x * 256 + threadIdx.x;   // over NB*NL*DI = 2^23
    const int d = idx & (DI - 1);
    const int l = (idx >> 9) & (NL - 1);
    const int b = idx >> 21;

    const float* ub = xz + (((size_t)(b * NL)) << 10);  // xz row stride 1024
    float acc = cb[d];
#pragma unroll
    for (int k = 0; k < 4; ++k) {
        const int ls = l - 3 + k;
        if (ls >= 0) acc += cw[d * 4 + k] * ub[(((size_t)ls) << 10) + d];
    }
    const float s = acc / (1.0f + __expf(-acc));        // SiLU
    u32[idx] = s;
    u16[idx] = (_Float16)s;
}

// ---------------------------------------------------------------------------
// delta = softplus(dt @ dt_proj_w^T + dt_proj_b); dt = x_dbl cols [0,16),
// x_dbl row stride padded to 64.
// ---------------------------------------------------------------------------
__global__ void __launch_bounds__(256)
dt_softplus_kernel(const float* __restrict__ xdbl,
                   const float* __restrict__ dtw,
                   const float* __restrict__ dtb,
                   float* __restrict__ delta)
{
    const int idx = blockIdx.x * 256 + threadIdx.x;   // over NM*DI = 2^23
    const int d = idx & (DI - 1);
    const int m = idx >> 9;
    const float* xr = xdbl + (size_t)m * 64;
    float acc = dtb[d];
#pragma unroll
    for (int r = 0; r < 16; ++r) acc += xr[r] * dtw[d * 16 + r];
    delta[idx] = (acc > 20.0f) ? acc : log1pf(__expf(acc));
}

// ---------------------------------------------------------------------------
// Selective scan: thread = one (b, d) channel, 16 f32 states in registers,
// sequential over L. Fused epilogue: y = (scan + u*D) * SiLU(z) -> f16.
// grid = NB * (DI/256) = 8 blocks x 256 threads.
// ---------------------------------------------------------------------------
__global__ void __launch_bounds__(256)
scan_kernel(const float* __restrict__ delta,
            const float* __restrict__ uact,
            const float* __restrict__ xdbl,
            const float* __restrict__ xz,
            const float* __restrict__ A_log,
            const float* __restrict__ Dp,
            _Float16* __restrict__ y16)
{
    const int b = blockIdx.x >> 1;
    const int d = ((blockIdx.x & 1) << 8) + threadIdx.x;

    float Ar[DSN];
#pragma unroll
    for (int s = 0; s < DSN; ++s) Ar[s] = -__expf(A_log[d * DSN + s]);
    const float Dd = Dp[d];

    float h[DSN];
#pragma unroll
    for (int s = 0; s < DSN; ++s) h[s] = 0.0f;

    for (int l = 0; l < NL; ++l) {
        const size_t m = (size_t)b * NL + l;
        const float dt = delta[m * DI + d];
        const float ut = uact[m * DI + d];

        // B_t, C_t broadcast across the block (uniform address -> cache hit)
        const float4* bp = (const float4*)(xdbl + m * 64 + 16);
        const float4* cp = (const float4*)(xdbl + m * 64 + 32);
        float Bv[DSN], Cv[DSN];
#pragma unroll
        for (int q = 0; q < 4; ++q) {
            const float4 bb = bp[q], cc = cp[q];
            Bv[4*q+0] = bb.x; Bv[4*q+1] = bb.y; Bv[4*q+2] = bb.z; Bv[4*q+3] = bb.w;
            Cv[4*q+0] = cc.x; Cv[4*q+1] = cc.y; Cv[4*q+2] = cc.z; Cv[4*q+3] = cc.w;
        }

        const float du = dt * ut;
        float y = 0.0f;
#pragma unroll
        for (int s = 0; s < DSN; ++s) {
            h[s] = __expf(dt * Ar[s]) * h[s] + du * Bv[s];
            y += h[s] * Cv[s];
        }

        const float zv = xz[(m << 10) + DI + d];
        const float yo = (y + ut * Dd) * (zv / (1.0f + __expf(-zv)));
        y16[m * DI + d] = (_Float16)yo;
    }
}

// ---------------------------------------------------------------------------
extern "C" void kernel_launch(void* const* d_in, const int* in_sizes, int n_in,
                              void* d_out, int out_size, void* d_ws, size_t ws_size,
                              hipStream_t stream)
{
    const float* x       = (const float*)d_in[0];
    const float* norm_w  = (const float*)d_in[1];
    const float* norm_b  = (const float*)d_in[2];
    const float* in_w    = (const float*)d_in[3];   // (1024, 256)
    const float* conv_w  = (const float*)d_in[4];   // (512, 1, 4)
    const float* conv_b  = (const float*)d_in[5];
    const float* xproj_w = (const float*)d_in[6];   // (48, 512)
    const float* dt_w    = (const float*)d_in[7];   // (512, 16)
    const float* dt_b    = (const float*)d_in[8];
    const float* A_log   = (const float*)d_in[9];   // (512, 16)
    const float* Dvec    = (const float*)d_in[10];  // (512,)
    const float* out_w   = (const float*)d_in[11];  // (256, 512)
    float* out = (float*)d_out;

    // bump allocator on workspace
    uint8_t* ws = (uint8_t*)d_ws;
    size_t off = 0;
    auto alloc = [&](size_t bytes) -> void* {
        void* p = (void*)(ws + off);
        off += (bytes + 255) & ~(size_t)255;
        return p;
    };

    _Float16* w_in16  = (_Float16*)alloc((size_t)1024 * 256 * 2);
    _Float16* w_xp16  = (_Float16*)alloc((size_t)64 * 512 * 2);     // padded 48->64
    _Float16* w_out16 = (_Float16*)alloc((size_t)256 * 512 * 2);
    _Float16* xn16    = (_Float16*)alloc((size_t)NM * NC * 2);
    float*    xz      = (float*)   alloc((size_t)NM * 1024 * 4);
    float*    uact    = (float*)   alloc((size_t)NM * DI * 4);
    _Float16* u16     = (_Float16*)alloc((size_t)NM * DI * 2);
    float*    xdbl    = (float*)   alloc((size_t)NM * 64 * 4);
    float*    delta   = (float*)   alloc((size_t)NM * DI * 4);
    _Float16* y16     = (_Float16*)alloc((size_t)NM * DI * 2);
    (void)ws_size; (void)in_sizes; (void)n_in; (void)out_size;

    // 1) weight conversions (f32 -> f16, x_proj padded to 64 rows)
    {
        int n = 1024 * 256;
        cvt_pad_f16_kernel<<<(n + 255) / 256, 256, 0, stream>>>(in_w, w_in16, 1024, 256, n);
        n = 64 * 512;
        cvt_pad_f16_kernel<<<(n + 255) / 256, 256, 0, stream>>>(xproj_w, w_xp16, 48, 512, n);
        n = 256 * 512;
        cvt_pad_f16_kernel<<<(n + 255) / 256, 256, 0, stream>>>(out_w, w_out16, 256, 512, n);
    }

    // 2) layernorm -> xn16 (M x 256, f16)
    layernorm_f16_kernel<<<NM, 256, 0, stream>>>(x, norm_w, norm_b, xn16);

    // 3) in_proj GEMM: xz (M x 1024) = xn16 @ w_in16^T   [WMMA]
    {
        dim3 grid(1024 / 64, NM / 128);
        wmma_gemm_f16f32_kernel<<<grid, 256, 0, stream>>>(xn16, w_in16, xz,
                                                          NM, 1024, 256, 0);
    }

    // 4) depthwise causal conv + SiLU -> uact (f32), u16 (f16)
    conv_silu_kernel<<<(NB * NL * DI) / 256, 256, 0, stream>>>(xz, conv_w, conv_b,
                                                               uact, u16);

    // 5) x_proj GEMM: xdbl (M x 64, cols 48..63 zero) = u16 @ w_xp16^T  [WMMA]
    {
        dim3 grid(64 / 64, NM / 128);
        wmma_gemm_f16f32_kernel<<<grid, 256, 0, stream>>>(u16, w_xp16, xdbl,
                                                          NM, 64, 512, 0);
    }

    // 6) delta = softplus(dt @ dt_proj_w^T + b)
    dt_softplus_kernel<<<(NM * DI) / 256, 256, 0, stream>>>(xdbl, dt_w, dt_b, delta);

    // 7) selective scan fused with skip + SiLU(z) gating -> y16
    scan_kernel<<<NB * 2, 256, 0, stream>>>(delta, uact, xdbl, xz, A_log, Dvec, y16);

    // 8) out_proj GEMM scattered into (B, C, H, W)  [WMMA]
    {
        dim3 grid(NC / 64, NM / 128);
        wmma_gemm_f16f32_kernel<<<grid, 256, 0, stream>>>(y16, w_out16, out,
                                                          NM, NC, 512, 1);
    }
}
